// MoELayer_74981539053721
// MI455X (gfx1250) — matrix operand order repaired and checked by reading, not compiled, not run on hardware
//
#include <hip/hip_runtime.h>
#include <hip/hip_bf16.h>

// ---------------------------------------------------------------------------
// MoE (8 experts, top-2, H=1024, FF=4096, 8192 tokens, fp32 in/out).
// Matrix-core bound (~0.41 TFLOP useful vs ~18us for one HBM weight sweep at
// 23.3 TB/s; 192MB L2 holds one expert's 48MB of weights + all of x).
// => bf16 WMMA (v_wmma_f32_16x16x32_bf16), fp32->bf16 convert while staging
// into LDS, double-buffered tiles (stage k+1 while WMMA on k), and async
// global->LDS DMA (ASYNCcnt) for the pure-copy A tile in pass 2.
//
// Workspace (~67.5 MB):
//   [0, 64MB)        h intermediate, bf16 [8192, 4096]
//   [64MB, +256KB)   combine weights, f32 [8192, 8]
//   [.., +64B)       loss accumulators: sum(logits^2), probs_sum[8]
// ---------------------------------------------------------------------------

#ifndef __has_builtin
#define __has_builtin(x) 0
#endif

#define NEXP 8
#define HDIM 1024
#define FDIM 4096
#define NTOK 8192            // 4 * 2048
#define BM   128             // token tile
#define BN   128             // output-column tile
#define BK   32              // k-step (== WMMA K for bf16)
#define KPAD 40              // padded K stride (bf16 elems) for transposed B tiles

typedef __bf16 bf16;
typedef bf16  v16bf __attribute__((ext_vector_type(16)));
typedef bf16  v8bf  __attribute__((ext_vector_type(8)));
typedef bf16  v2bf  __attribute__((ext_vector_type(2)));
typedef float v8f   __attribute__((ext_vector_type(8)));
typedef int   v4i   __attribute__((ext_vector_type(4)));

union V16 { v16bf v; v8bf h[2]; };

__device__ __forceinline__ unsigned short bfbits(float f) {
  unsigned u = __builtin_bit_cast(unsigned, f);
  u += 0x7FFFu + ((u >> 16) & 1u);            // round-to-nearest-even
  return (unsigned short)(u >> 16);
}
__device__ __forceinline__ bf16 f2bf(float f) {
  unsigned short s = bfbits(f);
  return __builtin_bit_cast(bf16, s);
}
// Pack two floats -> two bf16 in one u32 (low = first). Prefer the HW packed
// convert (v_cvt_pk_bf16_f32) when the toolchain exposes it.
#if __has_builtin(__builtin_amdgcn_cvt_pk_bf16_f32)
__device__ __forceinline__ unsigned f2bf2(float a, float b) {
  v2bf r = __builtin_amdgcn_cvt_pk_bf16_f32(a, b);
  return __builtin_bit_cast(unsigned, r);
}
#else
__device__ __forceinline__ unsigned f2bf2(float a, float b) {
  return (unsigned)bfbits(a) | ((unsigned)bfbits(b) << 16);
}
#endif

// Async global->LDS DMA (gfx1250 GLOBAL_LOAD_ASYNC_TO_LDS_B128, ASYNCcnt).
// Toolchain signature (from diagnostics): (v4i AS1*, v4i AS3*, imm, imm).
#if __has_builtin(__builtin_amdgcn_global_load_async_to_lds_b128) && \
    __has_builtin(__builtin_amdgcn_s_wait_asynccnt)
#define HAVE_ASYNC 1
#else
#define HAVE_ASYNC 0
#endif

__device__ __forceinline__ void copy16_g2l(const void* g, void* l) {
#if HAVE_ASYNC
  __builtin_amdgcn_global_load_async_to_lds_b128(
      (__attribute__((address_space(1))) v4i*)g,
      (__attribute__((address_space(3))) v4i*)l, 0, 0);
#else
  *(uint4*)l = *(const uint4*)g;
#endif
}
__device__ __forceinline__ void async_wait0() {
#if HAVE_ASYNC
  __builtin_amdgcn_s_wait_asynccnt(0);
#endif
}

// ---------------------------------------------------------------------------
// Router: logits -> softmax -> top2 -> renormalized combine weights [T, E].
// ---------------------------------------------------------------------------
__global__ void __launch_bounds__(256)
moe_router(const float* __restrict__ x, const float* __restrict__ wr,
           float* __restrict__ combine, float* __restrict__ lacc) {
  __shared__ float s_z;
  __shared__ float s_p[NEXP];
  int tid = threadIdx.x;
  if (tid == 0) s_z = 0.f;
  if (tid < NEXP) s_p[tid] = 0.f;
  __syncthreads();

  int t = blockIdx.x * blockDim.x + tid;
  float l[NEXP];
#pragma unroll
  for (int e = 0; e < NEXP; ++e) l[e] = 0.f;
  const float* xr = x + (size_t)t * HDIM;
  for (int k = 0; k < HDIM; ++k) {
    float xv = xr[k];
    const float4* w4 = (const float4*)(wr + k * NEXP);
    float4 a = w4[0], b = w4[1];
    l[0] += xv * a.x; l[1] += xv * a.y; l[2] += xv * a.z; l[3] += xv * a.w;
    l[4] += xv * b.x; l[5] += xv * b.y; l[6] += xv * b.z; l[7] += xv * b.w;
  }
  float zsum = 0.f, m = l[0];
#pragma unroll
  for (int e = 0; e < NEXP; ++e) { zsum += l[e] * l[e]; m = fmaxf(m, l[e]); }
  float p[NEXP], ps = 0.f;
#pragma unroll
  for (int e = 0; e < NEXP; ++e) { p[e] = __expf(l[e] - m); ps += p[e]; }
  float rps = __builtin_amdgcn_rcpf(ps);
#pragma unroll
  for (int e = 0; e < NEXP; ++e) p[e] *= rps;

  int i1 = 0; float p1 = p[0];
#pragma unroll
  for (int e = 1; e < NEXP; ++e) if (p[e] > p1) { p1 = p[e]; i1 = e; }
  int i2 = -1; float p2 = -1.f;
#pragma unroll
  for (int e = 0; e < NEXP; ++e) if (e != i1 && p[e] > p2) { p2 = p[e]; i2 = e; }
  float inv = __builtin_amdgcn_rcpf(p1 + p2);
#pragma unroll
  for (int e = 0; e < NEXP; ++e) {
    float c = (e == i1) ? p1 * inv : ((e == i2) ? p2 * inv : 0.f);
    combine[t * NEXP + e] = c;
  }

  atomicAdd(&s_z, zsum);
#pragma unroll
  for (int e = 0; e < NEXP; ++e) atomicAdd(&s_p[e], p[e]);
  __syncthreads();
  if (tid == 0) atomicAdd(&lacc[0], s_z);
  if (tid < NEXP) atomicAdd(&lacc[1 + tid], s_p[tid]);
}

// ---------------------------------------------------------------------------
// Pass 1 stager: A tile (x, packed u32 stores) + Wg/Wu tiles transposed [N][K].
// ---------------------------------------------------------------------------
__device__ __forceinline__ void stage_p1(const float* x, const float* wg,
                                         const float* wu, bf16* lA, bf16* lG,
                                         bf16* lU, int t0, int ff0, int k0,
                                         int tid) {
#pragma unroll
  for (int i = 0; i < 4; ++i) {
    int idx = (i * 256 + tid) * 4;
    int m = idx >> 5, k = idx & 31;
    float4 v = *(const float4*)(x + (size_t)(t0 + m) * HDIM + k0 + k);
    *(unsigned*)&lA[m * BK + k]     = f2bf2(v.x, v.y);
    *(unsigned*)&lA[m * BK + k + 2] = f2bf2(v.z, v.w);
  }
#pragma unroll
  for (int i = 0; i < 4; ++i) {
    int idx = (i * 256 + tid) * 4;
    int k = idx >> 7, n = idx & 127;
    float4 g = *(const float4*)(wg + (size_t)(k0 + k) * FDIM + ff0 + n);
    float4 u = *(const float4*)(wu + (size_t)(k0 + k) * FDIM + ff0 + n);
    lG[(n + 0) * KPAD + k] = f2bf(g.x); lG[(n + 1) * KPAD + k] = f2bf(g.y);
    lG[(n + 2) * KPAD + k] = f2bf(g.z); lG[(n + 3) * KPAD + k] = f2bf(g.w);
    lU[(n + 0) * KPAD + k] = f2bf(u.x); lU[(n + 1) * KPAD + k] = f2bf(u.y);
    lU[(n + 2) * KPAD + k] = f2bf(u.z); lU[(n + 3) * KPAD + k] = f2bf(u.w);
  }
}

// ---------------------------------------------------------------------------
// Pass 1 (per expert): h = silu(x @ Wg) * (x @ Wu)  -> bf16 [NTOK, FDIM].
// 8 waves (4x2), wave tile 32x64 for BOTH gate & up; double-buffered LDS.
// ---------------------------------------------------------------------------
__global__ void __launch_bounds__(256)
moe_gate_up(const float* __restrict__ x, const float* __restrict__ wg_all,
            const float* __restrict__ wu_all, bf16* __restrict__ hbuf, int e) {
  __shared__ __align__(16) bf16 ldsA[2][BM * BK];    // 2 x  8 KB
  __shared__ __align__(16) bf16 ldsG[2][BN * KPAD];  // 2 x 10 KB
  __shared__ __align__(16) bf16 ldsU[2][BN * KPAD];  // 2 x 10 KB

  const float* wg = wg_all + (size_t)e * HDIM * FDIM;
  const float* wu = wu_all + (size_t)e * HDIM * FDIM;
  int ff0 = blockIdx.x * BN;
  int t0  = blockIdx.y * BM;
  int tid  = threadIdx.x;
  int lane = tid & 31, wave = tid >> 5;
  int mbase = (wave >> 1) * 32, nbase = (wave & 1) * 64;
  int row15 = lane & 15;
  int kbA = (lane < 16) ? 0 : 16;   // A frag: hi lanes hold K=16..31
  int kbB = (lane < 16) ? 0 : 16;   // B frag: hi lanes hold K=16..31

  const v8f vzero = {0.f, 0.f, 0.f, 0.f, 0.f, 0.f, 0.f, 0.f};
  v8f accg[2][4], accu[2][4];
#pragma unroll
  for (int i = 0; i < 2; ++i)
#pragma unroll
    for (int j = 0; j < 4; ++j) { accg[i][j] = vzero; accu[i][j] = vzero; }

  stage_p1(x, wg, wu, ldsA[0], ldsG[0], ldsU[0], t0, ff0, 0, tid);
  __syncthreads();

  int cur = 0;
  for (int ks = 0; ks < HDIM / BK; ++ks) {
    int nxt = cur ^ 1;
    if (ks + 1 < HDIM / BK)
      stage_p1(x, wg, wu, ldsA[nxt], ldsG[nxt], ldsU[nxt], t0, ff0,
               (ks + 1) * BK, tid);

    V16 a[2];
#pragma unroll
    for (int mf = 0; mf < 2; ++mf) {
      int r = (mbase + mf * 16 + row15) * BK;
      a[mf].h[0] = *(const v8bf*)&ldsA[cur][r + kbA];
      a[mf].h[1] = *(const v8bf*)&ldsA[cur][r + kbA + 8];
    }
#pragma unroll
    for (int nf = 0; nf < 4; ++nf) {
      int c = (nbase + nf * 16 + row15) * KPAD;
      V16 bg, bu;
      bg.h[0] = *(const v8bf*)&ldsG[cur][c + kbB];
      bg.h[1] = *(const v8bf*)&ldsG[cur][c + kbB + 8];
      bu.h[0] = *(const v8bf*)&ldsU[cur][c + kbB];
      bu.h[1] = *(const v8bf*)&ldsU[cur][c + kbB + 8];
#pragma unroll
      for (int mf = 0; mf < 2; ++mf) {
        accg[mf][nf] = __builtin_amdgcn_wmma_f32_16x16x32_bf16(
            false, a[mf].v, false, bg.v, (short)0, accg[mf][nf], false, false);
        accu[mf][nf] = __builtin_amdgcn_wmma_f32_16x16x32_bf16(
            false, a[mf].v, false, bu.v, (short)0, accu[mf][nf], false, false);
      }
    }
    __syncthreads();
    cur = nxt;
  }

  // Epilogue: act = silu(g) * u = g * rcp(1+exp(-g)) * u  -> bf16 store.
  int moff = (lane < 16) ? 0 : 8;   // C/D layout: M = r (+8 for hi lanes)
#pragma unroll
  for (int mf = 0; mf < 2; ++mf)
#pragma unroll
    for (int nf = 0; nf < 4; ++nf)
#pragma unroll
      for (int r = 0; r < 8; ++r) {
        float g = accg[mf][nf][r], u = accu[mf][nf][r];
        float act = g * __builtin_amdgcn_rcpf(1.f + __expf(-g)) * u;
        int m = mbase + mf * 16 + moff + r;
        int n = nbase + nf * 16 + row15;
        hbuf[(size_t)(t0 + m) * FDIM + ff0 + n] = f2bf(act);
      }
}

// ---------------------------------------------------------------------------
// Pass 2 stager: A tile = straight bf16 copy (async DMA when available),
// B tile = Wd fp32 -> bf16 transposed [N][K].
// ---------------------------------------------------------------------------
__device__ __forceinline__ void stage_p2(const bf16* hbuf, const float* wd,
                                         bf16* lA, bf16* lB, int t0, int n0,
                                         int k0, int tid) {
#pragma unroll
  for (int i = 0; i < 2; ++i) {
    int idx = (i * 256 + tid) * 8;
    int m = idx >> 5, k = idx & 31;
    copy16_g2l(&hbuf[(size_t)(t0 + m) * FDIM + k0 + k], &lA[m * BK + k]);
  }
#pragma unroll
  for (int i = 0; i < 4; ++i) {
    int idx = (i * 256 + tid) * 4;
    int k = idx >> 7, n = idx & 127;
    float4 v = *(const float4*)(wd + (size_t)(k0 + k) * HDIM + n0 + n);
    lB[(n + 0) * KPAD + k] = f2bf(v.x); lB[(n + 1) * KPAD + k] = f2bf(v.y);
    lB[(n + 2) * KPAD + k] = f2bf(v.z); lB[(n + 3) * KPAD + k] = f2bf(v.w);
  }
}

// ---------------------------------------------------------------------------
// Pass 2 (per expert): out += combine[:,e] * (h @ Wd). Double-buffered LDS.
// Non-atomic RMW safe: disjoint tiles within a launch, experts serialized.
// ---------------------------------------------------------------------------
__global__ void __launch_bounds__(256)
moe_down(const bf16* __restrict__ hbuf, const float* __restrict__ wd_all,
         const float* __restrict__ combine, float* __restrict__ out, int e) {
  __shared__ __align__(16) bf16 ldsA[2][BM * BK];
  __shared__ __align__(16) bf16 ldsB[2][BN * KPAD];

  const float* wd = wd_all + (size_t)e * FDIM * HDIM;
  int n0 = blockIdx.x * BN;          // over HDIM
  int t0 = blockIdx.y * BM;
  int tid  = threadIdx.x;
  int lane = tid & 31, wave = tid >> 5;
  int mbase = (wave >> 1) * 32, nbase = (wave & 1) * 64;
  int row15 = lane & 15;
  int kbA = (lane < 16) ? 0 : 16;
  int kbB = (lane < 16) ? 0 : 16;

  const v8f vzero = {0.f, 0.f, 0.f, 0.f, 0.f, 0.f, 0.f, 0.f};
  v8f acc[2][4];
#pragma unroll
  for (int i = 0; i < 2; ++i)
#pragma unroll
    for (int j = 0; j < 4; ++j) acc[i][j] = vzero;

  stage_p2(hbuf, wd, ldsA[0], ldsB[0], t0, n0, 0, tid);
  async_wait0();
  __syncthreads();

  int cur = 0;
  for (int ks = 0; ks < FDIM / BK; ++ks) {
    int nxt = cur ^ 1;
    if (ks + 1 < FDIM / BK)
      stage_p2(hbuf, wd, ldsA[nxt], ldsB[nxt], t0, n0, (ks + 1) * BK, tid);

    V16 a[2];
#pragma unroll
    for (int mf = 0; mf < 2; ++mf) {
      int r = (mbase + mf * 16 + row15) * BK;
      a[mf].h[0] = *(const v8bf*)&ldsA[cur][r + kbA];
      a[mf].h[1] = *(const v8bf*)&ldsA[cur][r + kbA + 8];
    }
#pragma unroll
    for (int nf = 0; nf < 4; ++nf) {
      int c = (nbase + nf * 16 + row15) * KPAD;
      V16 b;
      b.h[0] = *(const v8bf*)&ldsB[cur][c + kbB];
      b.h[1] = *(const v8bf*)&ldsB[cur][c + kbB + 8];
#pragma unroll
      for (int mf = 0; mf < 2; ++mf)
        acc[mf][nf] = __builtin_amdgcn_wmma_f32_16x16x32_bf16(
            false, a[mf].v, false, b.v, (short)0, acc[mf][nf], false, false);
    }
    async_wait0();        // DMA for tile ks+1 must land before next compute
    __syncthreads();
    cur = nxt;
  }

  int moff = (lane < 16) ? 0 : 8;
#pragma unroll
  for (int mf = 0; mf < 2; ++mf)
#pragma unroll
    for (int nf = 0; nf < 4; ++nf)
#pragma unroll
      for (int r = 0; r < 8; ++r) {
        int m = mbase + mf * 16 + moff + r;
        int n = nbase + nf * 16 + row15;
        int t = t0 + m;
        float scale = combine[t * NEXP + e];
        out[(size_t)t * HDIM + n0 + n] += scale * acc[mf][nf][r];
      }
}

// ---------------------------------------------------------------------------
// Losses (matches reference's x8 accumulation).
// ---------------------------------------------------------------------------
__global__ void moe_losses(const float* __restrict__ lacc, float* __restrict__ out) {
  if (threadIdx.x == 0 && blockIdx.x == 0) {
    float z_single = lacc[0] / (float)(NTOK * NEXP);
    float aux = 0.f;
    for (int e = 0; e < NEXP; ++e) {
      float pm = lacc[1 + e] / (float)NTOK;
      aux += pm * __logf(pm + 1e-9f);   // == NEXP * mean_e(.)
    }
    out[(size_t)NTOK * HDIM + 0] = (float)NEXP * z_single;  // router_z_loss
    out[(size_t)NTOK * HDIM + 1] = (float)NEXP * aux;       // router_aux_loss
  }
}

extern "C" void kernel_launch(void* const* d_in, const int* in_sizes, int n_in,
                              void* d_out, int out_size, void* d_ws, size_t ws_size,
                              hipStream_t stream) {
  (void)in_sizes; (void)n_in; (void)ws_size;
  const float* x  = (const float*)d_in[0];
  const float* wr = (const float*)d_in[1];
  const float* wg = (const float*)d_in[2];
  const float* wu = (const float*)d_in[3];
  const float* wd = (const float*)d_in[4];
  float* out = (float*)d_out;

  char* ws = (char*)d_ws;
  bf16*  hbuf    = (bf16*)ws;                                   // 64 MB
  float* combine = (float*)(ws + (size_t)NTOK * FDIM * 2);      // 256 KB
  float* lacc    = combine + (size_t)NTOK * NEXP;               // 9 floats

  (void)hipMemsetAsync(out, 0, (size_t)out_size * sizeof(float), stream);
  (void)hipMemsetAsync(lacc, 0, 16 * sizeof(float), stream);

  moe_router<<<NTOK / 256, 256, 0, stream>>>(x, wr, combine, lacc);

  for (int e = 0; e < NEXP; ++e) {
    moe_gate_up<<<dim3(FDIM / BN, NTOK / BM), 256, 0, stream>>>(x, wg, wu, hbuf, e);
    moe_down   <<<dim3(HDIM / BN, NTOK / BM), 256, 0, stream>>>(hbuf, wd, combine, out, e);
  }

  moe_losses<<<1, 1, 0, stream>>>(lacc, out);
}
